// MTL_Decoder_37297495998701
// MI455X (gfx1250) — matrix operand order repaired and checked by reading, not compiled, not run on hardware
//
#include <hip/hip_runtime.h>
#include <hip/hip_bf16.h>
#include <math.h>
#include <stdint.h>

typedef __attribute__((ext_vector_type(16))) _Float16 v16h;
typedef __attribute__((ext_vector_type(8)))  _Float16 v8h;
typedef __attribute__((ext_vector_type(8)))  float    v8f;
typedef __attribute__((ext_vector_type(4)))  unsigned u32x4;
typedef __attribute__((ext_vector_type(8)))  int      i32x8;
typedef __attribute__((ext_vector_type(4)))  int      i32x4;

// GCC-style vector to match the async-LDS builtin's parameter type exactly
// (diagnostic showed: __attribute__((__vector_size__(4*sizeof(int)))) int __device__*)
typedef int gv4i __attribute__((vector_size(16)));
typedef __attribute__((address_space(1))) gv4i glb_v4i;
typedef __attribute__((address_space(3))) gv4i lds_v4i;

#define MPAD 112   // 100 rows padded to 7 tiles of 16
#define EMB  128
#define KCAT 160   // 132 padded to 5 chunks of 32

// ---- gfx1250 feature probes (guarded: fall back to sync copies) -------------
#if defined(__gfx1250__)
#if __has_builtin(__builtin_amdgcn_global_load_async_to_lds_b128) && \
    __has_builtin(__builtin_amdgcn_s_wait_asynccnt)
#define HAVE_ASYNC 1
#endif
#if __has_builtin(__builtin_amdgcn_tensor_load_to_lds) && \
    __has_builtin(__builtin_amdgcn_s_wait_tensorcnt)
#define HAVE_TDM 1
#endif
#endif
#ifndef HAVE_ASYNC
#define HAVE_ASYNC 0
#endif
#ifndef HAVE_TDM
#define HAVE_TDM 0
#endif
#if HAVE_TDM && __has_include(<hip/amd_detail/amd_gfx1250_TDM.h>)
#define TDM_SIX 1   // amdgpu-toolchain: 6-arg builtin
#else
#define TDM_SIX 0   // ROCm 7.2: 5-arg builtin
#endif

// 16B per-lane async copy global->LDS (ASYNCcnt) with sync fallback.
__device__ __forceinline__ void copy16_async(_Float16* dst_lds, const _Float16* src_glb) {
#if HAVE_ASYNC
  glb_v4i* g = (glb_v4i*)(unsigned long long)(uintptr_t)src_glb;
  lds_v4i* l = (lds_v4i*)(unsigned)(uintptr_t)dst_lds;  // low 32b = LDS offset
  __builtin_amdgcn_global_load_async_to_lds_b128(g, l, 0, 0);
#else
  *(v8h*)dst_lds = *(const v8h*)src_glb;
#endif
}
__device__ __forceinline__ void async_join() {
#if HAVE_ASYNC
  __builtin_amdgcn_s_wait_asynccnt(0);
#endif
}

// ---------------- WMMA fragment helpers (layouts per CDNA5 ISA) ---
__device__ __forceinline__ v16h load_a_frag(const _Float16* base, int ld) {
  int lane = threadIdx.x & 31;
  int row  = lane & 15;
  int ko   = (lane >> 4) * 8;
  const _Float16* p = base + row * ld + ko;
  v8h lo = *(const v8h*)(p);
  v8h hi = *(const v8h*)(p + 16);
  v16h a;
#pragma unroll
  for (int i = 0; i < 8; ++i) { a[i] = lo[i]; a[i + 8] = hi[i]; }
  return a;
}
__device__ __forceinline__ v16h load_bT_frag(const _Float16* baseT, int ld) {
  int lane = threadIdx.x & 31;
  int col  = lane & 15;
  int ko   = (lane >> 4) * 8;
  const _Float16* p = baseT + col * ld + ko;
  v8h lo = *(const v8h*)(p);
  v8h hi = *(const v8h*)(p + 16);
  v16h b;
#pragma unroll
  for (int i = 0; i < 8; ++i) { b[i] = lo[i]; b[i + 8] = hi[i]; }
  return b;
}
__device__ __forceinline__ v8f wmma32(v16h a, v16h b, v8f c) {
  return __builtin_amdgcn_wmma_f32_16x16x32_f16(false, a, false, b,
                                                (short)0, c, false, false);
}
__device__ __forceinline__ void store_c_f(float* base, int ld, v8f c) {
  int lane = threadIdx.x & 31;
  int col = lane & 15, r0 = (lane >> 4) * 8;
#pragma unroll
  for (int i = 0; i < 8; ++i) base[(r0 + i) * ld + col] = c[i];
}
__device__ __forceinline__ void store_c_h(_Float16* base, int ld, v8f c) {
  int lane = threadIdx.x & 31;
  int col = lane & 15, r0 = (lane >> 4) * 8;
#pragma unroll
  for (int i = 0; i < 8; ++i) base[(r0 + i) * ld + col] = (_Float16)c[i];
}
__device__ __forceinline__ void store_c_hT(_Float16* baseT, int ld, v8f c) {
  int lane = threadIdx.x & 31;
  int col = lane & 15, r0 = (lane >> 4) * 8;
#pragma unroll
  for (int i = 0; i < 8; ++i) baseT[col * ld + (r0 + i)] = (_Float16)c[i];
}

// ---------------- stage 0: weights -> f16, transposed (N-major), K padded ----
__global__ void wtrans_kernel(const float* __restrict__ src, _Float16* __restrict__ dstT,
                              int kin, int kpad, int nout) {
  int idx = blockIdx.x * blockDim.x + threadIdx.x;
  if (idx >= nout * kpad) return;
  int c = idx / kpad, r = idx - c * kpad;
  dstT[idx] = (_Float16)((r < kin) ? src[r * nout + c] : 0.f);
}

// ---------------- stage 0b: pack activations to f16, pad M and K -------------
__global__ void prep_acts_kernel(const float* __restrict__ nodes,
                                 const float* __restrict__ last,
                                 const float* __restrict__ attr,
                                 _Float16* __restrict__ nodesH,
                                 _Float16* __restrict__ catH) {
  int b = blockIdx.x;
  for (int i = threadIdx.x; i < MPAD * EMB; i += blockDim.x) {
    int r = i >> 7, c = i & 127;
    float v = (r < 100) ? nodes[(size_t)b * 100 * EMB + r * EMB + c] : 0.f;
    nodesH[(size_t)b * MPAD * EMB + i] = (_Float16)v;
  }
  for (int i = threadIdx.x; i < MPAD * KCAT; i += blockDim.x) {
    int r = i / KCAT, c = i - r * KCAT;
    float v = 0.f;
    if (r < 100) {
      if (c < 128)      v = last[(size_t)b * 100 * EMB + r * EMB + c];
      else if (c < 132) v = attr[(size_t)b * 100 * 4 + r * 4 + (c - 128)];
    }
    catH[(size_t)b * MPAD * KCAT + i] = (_Float16)v;
  }
}

// ---------------- stage 1: projection GEMM  O[b] = A[b](112xKpad) * WT -------
__global__ void proj_kernel(const _Float16* __restrict__ A,
                            const _Float16* __restrict__ WT,
                            _Float16* __restrict__ O, int Kpad, int storeT) {
  extern __shared__ __align__(16) char dynsm[];
  _Float16* As = (_Float16*)dynsm;        // 112 x Kpad (activations)
  _Float16* Ws = As + MPAD * Kpad;        // 128 x Kpad (weightT)
  int b = blockIdx.x;
  const _Float16* Ab = A + (size_t)b * MPAD * Kpad;
  int total = MPAD * Kpad;

#if HAVE_TDM
  // 2-D tile DMA: tensor = tile = MPAD x Kpad halves, row stride Kpad.
  if ((threadIdx.x >> 5) == 0) {          // one wave issues (TDM ignores EXEC)
    unsigned long long ga = (unsigned long long)(uintptr_t)Ab;
    unsigned la = (unsigned)(uintptr_t)As;  // low 32b of generic = LDS offset
    unsigned td0 = (unsigned)Kpad, td1 = MPAD;
    unsigned tl0 = (unsigned)Kpad, tl1 = MPAD;
    unsigned long long s0 = (unsigned long long)Kpad;
    u32x4 g0;
    g0[0] = 1u;                                                  // count=1
    g0[1] = la;                                                  // lds_addr
    g0[2] = (unsigned)(ga & 0xffffffffu);                        // gaddr[31:0]
    g0[3] = (unsigned)((ga >> 32) & 0x01ffffffu) | (2u << 30);   // gaddr[56:32] | type=2
    i32x8 g1;
    g1[0] = (int)(1u << 16);                                     // data_size=2B
    g1[1] = (int)((td0 & 0xffffu) << 16);                        // tensor_dim0[15:0]
    g1[2] = (int)((td0 >> 16) | ((td1 & 0xffffu) << 16));        // td0 hi | td1 lo
    g1[3] = (int)((td1 >> 16) | ((tl0 & 0xffffu) << 16));        // td1 hi | tile_dim0
    g1[4] = (int)(tl1 & 0xffffu);                                // tile_dim1, tile_dim2=0
    g1[5] = (int)(unsigned)(s0 & 0xffffffffu);                   // dim0_stride lo
    g1[6] = (int)(unsigned)((s0 >> 32) & 0xffffu);               // dim0_stride hi
    g1[7] = 0;
    i32x4 gz = {0, 0, 0, 0};
#if TDM_SIX
    i32x8 gz8 = {0, 0, 0, 0, 0, 0, 0, 0};
    __builtin_amdgcn_tensor_load_to_lds(g0, g1, gz, gz, gz8, 0);
#else
    __builtin_amdgcn_tensor_load_to_lds(g0, g1, gz, gz, 0);
#endif
    __builtin_amdgcn_s_wait_tensorcnt(0);
  }
#else
  for (int i = threadIdx.x * 8; i < total; i += blockDim.x * 8)
    copy16_async(As + i, Ab + i);
#endif
  // stage weightT (contiguous) via async-to-LDS
  for (int i = threadIdx.x * 8; i < EMB * Kpad; i += blockDim.x * 8)
    copy16_async(Ws + i, WT + i);
  async_join();
  __syncthreads();

  int w = threadIdx.x >> 5;
  int nk = Kpad >> 5;
  for (int t = w; t < 56; t += 8) {         // 7 Mtiles x 8 Ntiles
    int mt = t >> 3, nt = t & 7;
    v8f acc = {};
    for (int kc = 0; kc < nk; ++kc) {
      v16h a  = load_a_frag(As + mt * 16 * Kpad + kc * 32, Kpad);
      v16h bf = load_bT_frag(Ws + (nt * 16) * Kpad + kc * 32, Kpad);
      acc = wmma32(a, bf, acc);
    }
    if (!storeT)
      store_c_h(O + (size_t)b * MPAD * EMB + mt * 16 * EMB + nt * 16, EMB, acc);
    else  // V stored transposed: Vt[b][d=128][m=112]
      store_c_hT(O + (size_t)b * EMB * MPAD + (nt * 16) * MPAD + mt * 16, MPAD, acc);
  }
}

// ---------------- stage 2: attention for one (batch, head) -------------------
__global__ void attn_kernel(const _Float16* __restrict__ Q,
                            const _Float16* __restrict__ Km,
                            const _Float16* __restrict__ Vt,
                            const float* __restrict__ mask,
                            _Float16* __restrict__ O) {
  extern __shared__ __align__(16) char dynsm[];
  _Float16* Qh = (_Float16*)dynsm;      // 112 x 32 (K padded 16->32 with 0)
  _Float16* Kh = Qh + MPAD * 32;        // 112 x 32
  _Float16* Vh = Kh + MPAD * 32;        // 16 x 128 (m padded 112->128 with 0)
  float*    S  = (float*)(Vh + 16 * 128);      // 112 x 112 scores
  _Float16* P  = (_Float16*)(S + MPAD * MPAD); // 112 x 128 softmax weights

  int b = blockIdx.x, h = blockIdx.y;
  const _Float16* Qb = Q  + (size_t)b * MPAD * EMB + h * 16;
  const _Float16* Kb = Km + (size_t)b * MPAD * EMB + h * 16;
  const _Float16* Vb = Vt + (size_t)b * EMB * MPAD + (h * 16) * MPAD;

  for (int i = threadIdx.x; i < MPAD * 32; i += blockDim.x) {
    int r = i >> 5, c = i & 31;
    Qh[i] = (c < 16) ? Qb[r * EMB + c] : (_Float16)0.f;
    Kh[i] = (c < 16) ? Kb[r * EMB + c] : (_Float16)0.f;
  }
  for (int i = threadIdx.x; i < 16 * 128; i += blockDim.x) {
    int d = i >> 7, m = i & 127;
    Vh[i] = (m < MPAD) ? Vb[d * MPAD + m] : (_Float16)0.f;
  }
  __syncthreads();

  int w = threadIdx.x >> 5;
  for (int t = w; t < 49; t += 8) {           // 7x7 score tiles, K=32 (padded)
    int mt = t / 7, nt = t - mt * 7;
    v16h a  = load_a_frag(Qh + mt * 16 * 32, 32);
    v16h bf = load_bT_frag(Kh + nt * 16 * 32, 32);  // K row-major == BT of K^T
    v8f acc = {};
    acc = wmma32(a, bf, acc);
    store_c_f(S + mt * 16 * MPAD + nt * 16, MPAD, acc);
  }
  __syncthreads();

  int r = threadIdx.x;
  if (r < MPAD) {
    if (r < 100) {
      const float* mrow = mask + (size_t)b * 100 * 100 + r * 100;
      float mx = -1e30f;
      for (int m = 0; m < 100; ++m) {
        float v = S[r * MPAD + m] * 0.25f + mrow[m];  // 1/sqrt(16)
        S[r * MPAD + m] = v;
        mx = fmaxf(mx, v);
      }
      float sum = 0.f;
      for (int m = 0; m < 100; ++m) {
        float e = __expf(S[r * MPAD + m] - mx);
        S[r * MPAD + m] = e; sum += e;
      }
      float inv = 1.f / sum;
      for (int m = 0; m < 100; ++m) P[r * 128 + m] = (_Float16)(S[r * MPAD + m] * inv);
      for (int m = 100; m < 128; ++m) P[r * 128 + m] = (_Float16)0.f;
    } else {
      for (int m = 0; m < 128; ++m) P[r * 128 + m] = (_Float16)0.f;
    }
  }
  __syncthreads();

  if (w < 7) {                                // out = P(112x128) * V(128x16)
    int mt = w;
    v8f acc = {};
    for (int kc = 0; kc < 4; ++kc) {
      v16h a  = load_a_frag(P + mt * 16 * 128 + kc * 32, 128);
      v16h bf = load_bT_frag(Vh + kc * 32, 128);   // Vh is BT already
      acc = wmma32(a, bf, acc);
    }
    store_c_h(O + (size_t)b * MPAD * EMB + mt * 16 * EMB + h * 16, EMB, acc);
  }
}

// ---------------- stage 3: combine (Wc_s, Wc_t, gate fused) ------------------
__global__ void combine_kernel(const _Float16* __restrict__ Os,
                               const _Float16* __restrict__ Ot,
                               const _Float16* __restrict__ Cat,
                               const _Float16* __restrict__ WcsT,
                               const _Float16* __restrict__ WctT,
                               const _Float16* __restrict__ WgT,
                               const float* __restrict__ bcs,
                               const float* __restrict__ bct,
                               const float* __restrict__ bg,
                               _Float16* __restrict__ G) {
  extern __shared__ __align__(16) char dynsm[];
  _Float16* OsS  = (_Float16*)dynsm;        // 112x128
  _Float16* OtS  = OsS + MPAD * EMB;        // 112x128
  _Float16* CatS = OtS + MPAD * EMB;        // 112x160
  int b = blockIdx.x;
  const _Float16* p0 = Os  + (size_t)b * MPAD * EMB;
  const _Float16* p1 = Ot  + (size_t)b * MPAD * EMB;
  const _Float16* p2 = Cat + (size_t)b * MPAD * KCAT;
  for (int i = threadIdx.x * 8; i < MPAD * EMB; i += blockDim.x * 8) {
    copy16_async(OsS + i, p0 + i);
    copy16_async(OtS + i, p1 + i);
  }
  for (int i = threadIdx.x * 8; i < MPAD * KCAT; i += blockDim.x * 8)
    copy16_async(CatS + i, p2 + i);
  async_join();
  __syncthreads();

  int w = threadIdx.x >> 5, lane = threadIdx.x & 31;
  int col0 = lane & 15, r0 = (lane >> 4) * 8;
  for (int t = w; t < 56; t += 8) {
    int mt = t >> 3, nt = t & 7;
    v8f aS = {}, aT = {}, aG = {};
    for (int kc = 0; kc < 4; ++kc) {
      aS = wmma32(load_a_frag(OsS + mt * 16 * EMB + kc * 32, EMB),
                  load_bT_frag(WcsT + nt * 16 * EMB + kc * 32, EMB), aS);
      aT = wmma32(load_a_frag(OtS + mt * 16 * EMB + kc * 32, EMB),
                  load_bT_frag(WctT + nt * 16 * EMB + kc * 32, EMB), aT);
    }
    for (int kc = 0; kc < 5; ++kc)
      aG = wmma32(load_a_frag(CatS + mt * 16 * KCAT + kc * 32, KCAT),
                  load_bT_frag(WgT + nt * 16 * KCAT + kc * 32, KCAT), aG);
    int col = nt * 16 + col0;
    float bs = bcs[col], bt = bct[col], bgv = bg[col];
#pragma unroll
    for (int i = 0; i < 8; ++i) {
      float g = 1.f / (1.f + __expf(-(aG[i] + bgv)));
      float val = g * (aS[i] + bs) + (1.f - g) * (aT[i] + bt);
      G[(size_t)b * MPAD * EMB + (mt * 16 + r0 + i) * EMB + col] = (_Float16)val;
    }
  }
}

// ---------------- stage 4: pointer score + clip-tanh softmax -----------------
__global__ void pointer_kernel(const _Float16* __restrict__ G,
                               const _Float16* __restrict__ Nh,
                               const float* __restrict__ mask,
                               float* __restrict__ probs) {
  extern __shared__ __align__(16) char dynsm[];
  _Float16* Gs = (_Float16*)dynsm;     // 112x128
  _Float16* Ns = Gs + MPAD * EMB;      // 112x128 (row-major nodes == BT of nodes^T)
  float*    S  = (float*)(Ns + MPAD * EMB);  // 112x112
  int b = blockIdx.x;
  const _Float16* pg = G  + (size_t)b * MPAD * EMB;
  const _Float16* pn = Nh + (size_t)b * MPAD * EMB;
  for (int i = threadIdx.x * 8; i < MPAD * EMB; i += blockDim.x * 8) {
    copy16_async(Gs + i, pg + i);
    copy16_async(Ns + i, pn + i);
  }
  async_join();
  __syncthreads();

  int w = threadIdx.x >> 5;
  for (int t = w; t < 49; t += 8) {
    int mt = t / 7, nt = t - mt * 7;
    v8f acc = {};
    for (int kc = 0; kc < 4; ++kc) {
      v16h a  = load_a_frag(Gs + mt * 16 * EMB + kc * 32, EMB);
      v16h bf = load_bT_frag(Ns + nt * 16 * EMB + kc * 32, EMB);
      acc = wmma32(a, bf, acc);
    }
    store_c_f(S + mt * 16 * MPAD + nt * 16, MPAD, acc);
  }
  __syncthreads();

  int r = threadIdx.x;
  if (r < 100) {
    const float* mrow = mask + (size_t)b * 100 * 100 + r * 100;
    float mx = -1e30f;
    for (int m = 0; m < 100; ++m) {
      float v = 10.f * tanhf(S[r * MPAD + m] * 0.0883883476483184f) + mrow[m];
      S[r * MPAD + m] = v;
      mx = fmaxf(mx, v);
    }
    float sum = 0.f;
    for (int m = 0; m < 100; ++m) {
      float e = __expf(S[r * MPAD + m] - mx);
      S[r * MPAD + m] = e; sum += e;
    }
    float inv = 1.f / sum;
    float* orow = probs + (size_t)b * 100 * 100 + (size_t)r * 100;
    for (int m = 0; m < 100; ++m) orow[m] = S[r * MPAD + m] * inv;
  }
}

// ---------------- host: orchestration ---------------------------------------
extern "C" void kernel_launch(void* const* d_in, const int* in_sizes, int n_in,
                              void* d_out, int out_size, void* d_ws, size_t ws_size,
                              hipStream_t stream) {
  (void)n_in; (void)out_size; (void)ws_size;
  const float* nodes = (const float*)d_in[0];
  const float* last  = (const float*)d_in[1];
  const float* attr  = (const float*)d_in[2];
  const float* mask  = (const float*)d_in[3];
  const float* Wq_s  = (const float*)d_in[4];
  const float* Wk_s  = (const float*)d_in[5];
  const float* Wv_s  = (const float*)d_in[6];
  const float* Wq_t  = (const float*)d_in[7];
  const float* Wk_t  = (const float*)d_in[8];
  const float* Wv_t  = (const float*)d_in[9];
  const float* Wc_s  = (const float*)d_in[10];
  const float* bc_s  = (const float*)d_in[11];
  const float* Wc_t  = (const float*)d_in[12];
  const float* bc_t  = (const float*)d_in[13];
  const float* Wg    = (const float*)d_in[14];
  const float* bg    = (const float*)d_in[15];

  const int B = in_sizes[0] / (100 * EMB);

  _Float16* wsh = (_Float16*)d_ws;
  size_t off = 0;
  auto take = [&](size_t n) {
    _Float16* p = wsh + off;
    off += (n + 15) & ~(size_t)15;
    return p;
  };
  _Float16* nodesH = take((size_t)B * MPAD * EMB);
  _Float16* catH   = take((size_t)B * MPAD * KCAT);
  _Float16* WqsT = take((size_t)EMB * KCAT);
  _Float16* WqtT = take((size_t)EMB * KCAT);
  _Float16* WgT  = take((size_t)EMB * KCAT);
  _Float16* WksT = take((size_t)EMB * EMB);
  _Float16* WvsT = take((size_t)EMB * EMB);
  _Float16* WktT = take((size_t)EMB * EMB);
  _Float16* WvtT = take((size_t)EMB * EMB);
  _Float16* WcsT = take((size_t)EMB * EMB);
  _Float16* WctT = take((size_t)EMB * EMB);
  _Float16* Ks  = take((size_t)B * MPAD * EMB);
  _Float16* Kt  = take((size_t)B * MPAD * EMB);
  _Float16* VsT = take((size_t)B * EMB * MPAD);
  _Float16* VtT = take((size_t)B * EMB * MPAD);
  _Float16* Qs  = take((size_t)B * MPAD * EMB);
  _Float16* Qt  = take((size_t)B * MPAD * EMB);
  _Float16* Os  = take((size_t)B * MPAD * EMB);
  _Float16* Ot  = take((size_t)B * MPAD * EMB);
  _Float16* Gg  = take((size_t)B * MPAD * EMB);

  const int g160 = (EMB * KCAT + 255) / 256;
  const int g128 = (EMB * EMB + 255) / 256;
  wtrans_kernel<<<g160, 256, 0, stream>>>(Wq_s, WqsT, 132, KCAT, EMB);
  wtrans_kernel<<<g160, 256, 0, stream>>>(Wq_t, WqtT, 132, KCAT, EMB);
  wtrans_kernel<<<g160, 256, 0, stream>>>(Wg,   WgT,  132, KCAT, EMB);
  wtrans_kernel<<<g128, 256, 0, stream>>>(Wk_s, WksT, EMB, EMB, EMB);
  wtrans_kernel<<<g128, 256, 0, stream>>>(Wv_s, WvsT, EMB, EMB, EMB);
  wtrans_kernel<<<g128, 256, 0, stream>>>(Wk_t, WktT, EMB, EMB, EMB);
  wtrans_kernel<<<g128, 256, 0, stream>>>(Wv_t, WvtT, EMB, EMB, EMB);
  wtrans_kernel<<<g128, 256, 0, stream>>>(Wc_s, WcsT, EMB, EMB, EMB);
  wtrans_kernel<<<g128, 256, 0, stream>>>(Wc_t, WctT, EMB, EMB, EMB);

  prep_acts_kernel<<<B, 256, 0, stream>>>(nodes, last, attr, nodesH, catH);

  const size_t sh128 = (size_t)(MPAD + EMB) * EMB * sizeof(_Float16);
  const size_t sh160 = (size_t)(MPAD + EMB) * KCAT * sizeof(_Float16);
  proj_kernel<<<B, 256, sh160, stream>>>(catH,   WqsT, Qs,  KCAT, 0);
  proj_kernel<<<B, 256, sh160, stream>>>(catH,   WqtT, Qt,  KCAT, 0);
  proj_kernel<<<B, 256, sh128, stream>>>(nodesH, WksT, Ks,  EMB, 0);
  proj_kernel<<<B, 256, sh128, stream>>>(nodesH, WktT, Kt,  EMB, 0);
  proj_kernel<<<B, 256, sh128, stream>>>(nodesH, WvsT, VsT, EMB, 1);
  proj_kernel<<<B, 256, sh128, stream>>>(nodesH, WvtT, VtT, EMB, 1);

  const size_t shAttn = (size_t)(MPAD * 32 * 2 + 16 * 128) * sizeof(_Float16) +
                        (size_t)MPAD * MPAD * sizeof(float) +
                        (size_t)MPAD * 128 * sizeof(_Float16);
  attn_kernel<<<dim3(B, 8), 256, shAttn, stream>>>(Qs, Ks, VsT, mask, Os);
  attn_kernel<<<dim3(B, 8), 256, shAttn, stream>>>(Qt, Kt, VtT, mask, Ot);

  const size_t shComb = (size_t)(MPAD * EMB * 2 + MPAD * KCAT) * sizeof(_Float16);
  combine_kernel<<<B, 256, shComb, stream>>>(Os, Ot, catH, WcsT, WctT, WgT,
                                             bc_s, bc_t, bg, Gg);

  const size_t shPtr = (size_t)(MPAD * EMB * 2) * sizeof(_Float16) +
                       (size_t)MPAD * MPAD * sizeof(float);
  pointer_kernel<<<B, 256, shPtr, stream>>>(Gg, nodesH, mask, (float*)d_out);
}
